// TMatching_79216376807993
// MI455X (gfx1250) — matrix operand order repaired and checked by reading, not compiled, not run on hardware
//
#include <hip/hip_runtime.h>
#include <hip/hip_bf16.h>

#define G_  128
#define N_  131072LL
#define E_  262144LL
#define D_  256
#define DEI 64
#define DE2 128

typedef __attribute__((ext_vector_type(16))) __bf16 v16bf;
typedef __attribute__((ext_vector_type(8)))  float  v8f;

union FragBF { v16bf v; unsigned int u[8]; };
union FragF  { v8f v; float f[8]; };

__device__ __forceinline__ unsigned short f2bf(float x) {
  unsigned int u = __float_as_uint(x);
  u += 0x7fffu + ((u >> 16) & 1u);
  return (unsigned short)(u >> 16);
}

// ---- gemm flags (compile-time template parameter) ----
#define GF_RELU 1    // relu epilogue
#define GF_ACC  2    // accumulate into existing C
#define GF_OPPN 4    // B row base offset to opposite 512-block (scores: q @ k_opp^T)
#define GF_OPPK 8    // B k offset to opposite 512-block (attend: probs @ v_opp)
#define GF_IH   16   // A/C rows = H blocks (row r -> (r/512)*1024 + r%512)
#define GF_IG   32   // A/C rows = G blocks (+512)
#define GF_CBT  64   // write bf16 mirror transposed: Cb[col*ldcb + row]

template <int FLAGS>
__device__ __forceinline__ long long maprow(int r) {
  if (FLAGS & GF_IH) return (((long long)(r >> 9)) << 10) + (r & 511);
  if (FLAGS & GF_IG) return (((long long)(r >> 9)) << 10) + (r & 511) + 512;
  return (long long)r;
}

// async global -> LDS 128-bit copy (CDNA5 ASYNCcnt path)
__device__ __forceinline__ void async_ld_b128(unsigned int lds_off, const void* gaddr) {
  asm volatile("global_load_async_to_lds_b128 %0, %1, off"
               :: "v"(lds_off), "v"(gaddr) : "memory");
}
__device__ __forceinline__ void wait_async0() {
  asm volatile("s_wait_asynccnt 0x0" ::: "memory");
}

// C[M,Nc] (+)= act(A[M,K] @ B^T + bias), with B stored [Nc][K] (row n = output col).
// block = 256 threads (8 waves). Block tile 128x128. K-step 32.
// wave w: rows rg*32..+32 (rg=w>>1), cols cg*64..+64 (cg=w&1) => 8 wmma / K-step.
// Double-buffered LDS, filled with global_load_async_to_lds_b128.
template <int FLAGS>
__global__ __launch_bounds__(256) void gemm_bf16_kernel(
    const unsigned short* __restrict__ A, int lda,
    const unsigned short* __restrict__ B, int ldb,
    float* __restrict__ C, int ldc,
    unsigned short* __restrict__ Cb, int ldcb,
    const float* __restrict__ bias, int K) {
  __shared__ __align__(16) unsigned short lds_a[2][128 * 32];
  __shared__ __align__(16) unsigned short lds_b[2][128 * 32];   // [n][k]

  const int tid  = threadIdx.x;
  const int wave = tid >> 5;
  const int lane = tid & 31;
  const int hl   = lane >> 4;
  const int ll   = lane & 15;
  const int rg   = wave >> 1;       // row group 0..3
  const int cg   = wave & 1;        // col group 0..1
  const int mb   = blockIdx.x * 128;
  const int n0   = blockIdx.y * 128;

  const unsigned short* Beff = B;
  if (FLAGS & (GF_OPPN | GF_OPPK)) {
    long long blk = maprow<FLAGS>(mb) >> 9;
    long long opb = ((blk >> 1) << 10) + ((blk & 1) ? 0 : 512);
    Beff = B + ((FLAGS & GF_OPPN) ? opb * (long long)ldb : opb);
  }

  const int srow = tid >> 2, sseg = tid & 3;     // staging coords (2 chunks/thread)
  auto stage = [&](int k0, int buf) {
#pragma unroll
    for (int c = 0; c < 2; ++c) {
      int row = srow + c * 64;
      long long gr = maprow<FLAGS>(mb + row);
      async_ld_b128((unsigned int)(size_t)&lds_a[buf][row * 32 + sseg * 8],
                    A + gr * (long long)lda + k0 + sseg * 8);
      async_ld_b128((unsigned int)(size_t)&lds_b[buf][row * 32 + sseg * 8],
                    Beff + (long long)(n0 + row) * ldb + k0 + sseg * 8);
    }
  };

  stage(0, 0);   // prologue fill overlaps accumulator init

  FragF acc[2][4];
  if (FLAGS & GF_ACC) {
#pragma unroll
    for (int s = 0; s < 2; ++s)
#pragma unroll
      for (int t = 0; t < 4; ++t)
#pragma unroll
        for (int r = 0; r < 8; ++r) {
          long long grow = maprow<FLAGS>(mb + rg * 32 + s * 16 + r + 8 * hl);
          acc[s][t].f[r] = C[grow * ldc + n0 + cg * 64 + t * 16 + ll];
        }
  } else {
#pragma unroll
    for (int s = 0; s < 2; ++s)
#pragma unroll
      for (int t = 0; t < 4; ++t)
#pragma unroll
        for (int r = 0; r < 8; ++r) acc[s][t].f[r] = 0.0f;
  }

  int cur = 0;
  for (int k0 = 0; k0 < K; k0 += 32) {
    wait_async0();        // my async fills of buf[cur] done
    __syncthreads();      // everyone's fills visible; prior reads of buf[cur^1] done
    if (k0 + 32 < K) stage(k0 + 32, cur ^ 1);

    FragBF af[2];
#pragma unroll
    for (int s = 0; s < 2; ++s) {
      const unsigned short* ar = &lds_a[cur][(rg * 32 + s * 16 + ll) * 32 + hl * 8];
      af[s].u[0] = *(const unsigned int*)(ar + 0);
      af[s].u[1] = *(const unsigned int*)(ar + 2);
      af[s].u[2] = *(const unsigned int*)(ar + 4);
      af[s].u[3] = *(const unsigned int*)(ar + 6);
      af[s].u[4] = *(const unsigned int*)(ar + 16);
      af[s].u[5] = *(const unsigned int*)(ar + 18);
      af[s].u[6] = *(const unsigned int*)(ar + 20);
      af[s].u[7] = *(const unsigned int*)(ar + 22);
    }
#pragma unroll
    for (int t = 0; t < 4; ++t) {
      FragBF bf;
      const unsigned short* br = &lds_b[cur][(cg * 64 + t * 16 + ll) * 32 + hl * 8];
#pragma unroll
      for (int j = 0; j < 8; ++j) bf.u[j] = *(const unsigned int*)(br + 2 * j + (j >= 4 ? 8 : 0));
      acc[0][t].v = __builtin_amdgcn_wmma_f32_16x16x32_bf16(
          false, af[0].v, false, bf.v, (short)0, acc[0][t].v, false, false);
      acc[1][t].v = __builtin_amdgcn_wmma_f32_16x16x32_bf16(
          false, af[1].v, false, bf.v, (short)0, acc[1][t].v, false, false);
    }
    cur ^= 1;
  }

#pragma unroll
  for (int s = 0; s < 2; ++s)
#pragma unroll
    for (int t = 0; t < 4; ++t) {
      int col = n0 + cg * 64 + t * 16 + ll;
      float bv = bias ? bias[col] : 0.0f;
#pragma unroll
      for (int r = 0; r < 8; ++r) {
        long long grow = maprow<FLAGS>(mb + rg * 32 + s * 16 + r + 8 * hl);
        float v = acc[s][t].f[r] + bv;
        if (FLAGS & GF_RELU) v = fmaxf(v, 0.0f);
        C[grow * ldc + col] = v;
        if (Cb) {
          if (FLAGS & GF_CBT) Cb[(long long)col * ldcb + grow] = f2bf(v);
          else                Cb[grow * (long long)ldcb + col] = f2bf(v);
        }
      }
    }
}

// ---------------- elementwise / scatter / reduce kernels ----------------

__global__ void cast_bf16_kernel(const float* __restrict__ s,
                                 unsigned short* __restrict__ d, long long n) {
  long long i = (long long)blockIdx.x * 256 + threadIdx.x;
  if (i < n) d[i] = f2bf(s[i]);
}

// transposed cast: dst[n*K+k] = bf16(src[k*Nc+n])  (weights -> [n][k] layout)
__global__ void cast_t_kernel(const float* __restrict__ src,
                              unsigned short* __restrict__ dst, int K, int Nc) {
  long long i = (long long)blockIdx.x * 256 + threadIdx.x;
  if (i < (long long)K * Nc) {
    int n = (int)(i / K), k = (int)(i % K);
    dst[i] = f2bf(src[(long long)k * Nc + n]);
  }
}

__global__ void copy_cast_kernel(const float* __restrict__ s, float* __restrict__ d,
                                 unsigned short* __restrict__ db, long long n) {
  long long i = (long long)blockIdx.x * 256 + threadIdx.x;
  if (i < n) { float v = s[i]; d[i] = v; db[i] = f2bf(v); }
}

__global__ void copy_kernel(const float* __restrict__ s, float* __restrict__ d, long long n) {
  long long i = (long long)blockIdx.x * 256 + threadIdx.x;
  if (i < n) d[i] = s[i];
}

__global__ void efscatter_kernel(const float* __restrict__ ef, const int* __restrict__ dst,
                                 float* __restrict__ efsum, float* __restrict__ cnt) {
  long long e = blockIdx.x; int d = threadIdx.x;
  int v = dst[e];
  atomicAdd(&efsum[(long long)v * DE2 + d], ef[e * DE2 + d]);
  if (d == 0) atomicAdd(&cnt[v], 1.0f);
}

__global__ void yscatter_kernel(const float* __restrict__ y, const int* __restrict__ src,
                                const int* __restrict__ dst, float* __restrict__ acc) {
  long long e = blockIdx.x; int d = threadIdx.x;
  int s = src[e], v = dst[e];
  atomicAdd(&acc[(long long)v * D_ + d], y[(long long)s * D_ + d]);
}

__global__ void conv_fin_kernel(const float* __restrict__ acc, const float* __restrict__ cnt,
                                const float* __restrict__ b, float* __restrict__ out) {
  long long i = (long long)blockIdx.x * 256 + threadIdx.x;
  int d = (int)(i & 255); long long r = i >> 8;
  float c = cnt[r];
  float bb = (c > 0.0f) ? b[d] : 0.0f;
  float v = acc[i] / fmaxf(c, 1.0f) + bb;
  out[i] = fmaxf(v, 0.0f);
}

__global__ __launch_bounds__(256) void gnorm_kernel(
    const float* __restrict__ x, const float* __restrict__ w, const float* __restrict__ b,
    const float* __restrict__ ms, float* __restrict__ out, unsigned short* __restrict__ outbf) {
  int s = blockIdx.x, d = threadIdx.x;
  long long base = (long long)s * 512 * D_;
  float sum = 0.0f;
  for (int r = 0; r < 512; ++r) sum += x[base + (long long)r * D_ + d];
  float mean = sum * (1.0f / 512.0f);
  float msd = ms[d];
  float vs = 0.0f;
  for (int r = 0; r < 512; ++r) {
    float o = x[base + (long long)r * D_ + d] - msd * mean;
    vs += o * o;
  }
  float scale = w[d] * rsqrtf(vs * (1.0f / 512.0f) + 1e-5f);
  float bd = b[d];
  for (int r = 0; r < 512; ++r) {
    long long idx = base + (long long)r * D_ + d;
    float y = scale * (x[idx] - msd * mean) + bd;
    out[idx] = y;
    if (outbf) outbf[idx] = f2bf(y);
  }
}

__global__ __launch_bounds__(256) void softmax_kernel(const float* __restrict__ scores,
                                                      unsigned short* __restrict__ probs) {
  __shared__ float red[256];
  long long r = blockIdx.x; int t = threadIdx.x;
  const float* row = scores + r * 512;
  float a = row[t], b = row[t + 256];
  red[t] = fmaxf(a, b); __syncthreads();
  for (int o = 128; o > 0; o >>= 1) { if (t < o) red[t] = fmaxf(red[t], red[t + o]); __syncthreads(); }
  float mx = red[0]; __syncthreads();
  float ea = __expf(a - mx), eb = __expf(b - mx);
  red[t] = ea + eb; __syncthreads();
  for (int o = 128; o > 0; o >>= 1) { if (t < o) red[t] += red[t + o]; __syncthreads(); }
  float inv = 1.0f / red[0];
  probs[r * 512 + t]       = f2bf(ea * inv);
  probs[r * 512 + t + 256] = f2bf(eb * inv);
}

__global__ void mix_kernel(float* __restrict__ xcur, const float* __restrict__ xn,
                           unsigned short* __restrict__ xbf, long long n) {
  long long i = (long long)blockIdx.x * 256 + threadIdx.x;
  if (i < n) { float v = xcur[i] * 0.5f + xn[i] * 0.5f; xcur[i] = v; xbf[i] = f2bf(v); }
}

__global__ void segmean_kernel(const float* __restrict__ x, float* __restrict__ xout) {
  int s = blockIdx.x, d = threadIdx.x;
  long long base = (long long)s * 512 * D_;
  float sum = 0.0f;
  for (int r = 0; r < 512; ++r) sum += x[base + (long long)r * D_ + d];
  xout[(long long)s * D_ + d] = sum * (1.0f / 512.0f);
}

__global__ __launch_bounds__(256) void final_kernel(
    const float* __restrict__ xout, const float* __restrict__ f1W, const float* __restrict__ f1b,
    const float* __restrict__ f2W, const float* __restrict__ f2b,
    float* __restrict__ dscores, float* __restrict__ dx1, float* __restrict__ dx2) {
  __shared__ float xs[256];
  __shared__ float red[256];
  int g = blockIdx.x, t = threadIdx.x;
  float a = xout[(long long)(2 * g) * D_ + t];
  float b = xout[(long long)(2 * g + 1) * D_ + t];
  dx1[(long long)g * D_ + t] = a;
  dx2[(long long)g * D_ + t] = b;
  xs[t] = a - b; __syncthreads();
  float h = f1b[t];
  for (int d = 0; d < 256; ++d) h += xs[d] * f1W[d * 256 + t];
  h = fmaxf(h, 0.0f);
  red[t] = h * f2W[t * 2 + 0]; __syncthreads();
  for (int o = 128; o > 0; o >>= 1) { if (t < o) red[t] += red[t + o]; __syncthreads(); }
  float s0 = red[0] + f2b[0]; __syncthreads();
  red[t] = h * f2W[t * 2 + 1]; __syncthreads();
  for (int o = 128; o > 0; o >>= 1) { if (t < o) red[t] += red[t + o]; __syncthreads(); }
  float s1 = red[0] + f2b[1];
  if (t == 0) {
    float m = fmaxf(s0, s1);
    float e0 = __expf(s0 - m), e1 = __expf(s1 - m);
    float inv = 1.0f / (e0 + e1);
    dscores[2 * g + 0] = e0 * inv;
    dscores[2 * g + 1] = e1 * inv;
  }
}

// ---------------------------- host orchestration ----------------------------

extern "C" void kernel_launch(void* const* d_in, const int* in_sizes, int n_in,
                              void* d_out, int out_size, void* d_ws, size_t ws_size,
                              hipStream_t stream) {
  (void)in_sizes; (void)n_in; (void)out_size; (void)ws_size;
  const float* x_in   = (const float*)d_in[0];
  const float* efeat  = (const float*)d_in[1];
  const int*   eidx   = (const int*)d_in[2];
  const float* ef_W   = (const float*)d_in[4];
  const float* ef_b   = (const float*)d_in[5];
  const float* fin1_W = (const float*)d_in[6];
  const float* fin1_b = (const float*)d_in[7];
  const float* fin2_W = (const float*)d_in[8];
  const float* fin2_b = (const float*)d_in[9];
  const float* gn_w   = (const float*)d_in[10];
  const float* gn_b   = (const float*)d_in[11];
  const float* gn_ms  = (const float*)d_in[12];
  const float* hK = (const float*)d_in[13];
  const float* hQ = (const float*)d_in[14];
  const float* hV = (const float*)d_in[15];
  const float* gK = (const float*)d_in[16];
  const float* gQ = (const float*)d_in[17];
  const float* gV = (const float*)d_in[18];
  const float* last_W = (const float*)d_in[19];
  const float* last_b = (const float*)d_in[20];
  const float* conv_W[4] = {(const float*)d_in[21], (const float*)d_in[23],
                            (const float*)d_in[25], (const float*)d_in[27]};
  const float* conv_b[4] = {(const float*)d_in[22], (const float*)d_in[24],
                            (const float*)d_in[26], (const float*)d_in[28]};
  const int* esrc = eidx;
  const int* edst = eidx + E_;

  char* p = (char*)d_ws;
  auto alloc = [&](size_t bytes) -> char* {
    char* r = p; p += (bytes + 255) & ~(size_t)255; return r;
  };
  float* xcur    = (float*)alloc(N_ * D_ * 4);
  float* xnew    = (float*)alloc(N_ * D_ * 4);
  float* accb    = (float*)alloc(N_ * D_ * 4);
  float* ybuf    = (float*)alloc(N_ * D_ * 4);
  float* ctmp    = (float*)alloc(N_ * D_ * 4);
  float* xcross  = (float*)alloc(N_ * D_ * 4);
  float* xcrossn = (float*)alloc(N_ * D_ * 4);
  float* scores  = (float*)alloc(N_ * 512 * 4);
  float* efbuf   = (float*)alloc(E_ * DE2 * 4);
  float* efsum   = (float*)alloc(N_ * DE2 * 4);
  float* cnt     = (float*)alloc(N_ * 4);
  float* xout    = (float*)alloc(256 * 256 * 4);
  unsigned short* xbf     = (unsigned short*)alloc(N_ * D_ * 2);
  unsigned short* xnewbf  = (unsigned short*)alloc(N_ * D_ * 2);
  unsigned short* qbf     = (unsigned short*)alloc(N_ * D_ * 2);
  unsigned short* kbf     = (unsigned short*)alloc(N_ * D_ * 2);
  unsigned short* vT      = (unsigned short*)alloc(N_ * D_ * 2);  // [d][node]
  unsigned short* xcnbf   = (unsigned short*)alloc(N_ * D_ * 2);
  unsigned short* probs   = (unsigned short*)alloc(N_ * 512 * 2);
  unsigned short* efsumbf = (unsigned short*)alloc(N_ * DE2 * 2);
  unsigned short* ebf     = (unsigned short*)alloc(E_ * DEI * 2);
  // transposed bf16 weights ([n][k])
  unsigned short* wEfT   = (unsigned short*)alloc(DE2 * DEI * 2);
  unsigned short* wHKT   = (unsigned short*)alloc(65536 * 2);
  unsigned short* wHQT   = (unsigned short*)alloc(65536 * 2);
  unsigned short* wHVT   = (unsigned short*)alloc(65536 * 2);
  unsigned short* wGKT   = (unsigned short*)alloc(65536 * 2);
  unsigned short* wGQT   = (unsigned short*)alloc(65536 * 2);
  unsigned short* wGVT   = (unsigned short*)alloc(65536 * 2);
  unsigned short* wLaTT  = (unsigned short*)alloc(65536 * 2);    // last top^T
  unsigned short* wLaBT  = (unsigned short*)alloc(65536 * 2);    // last bot^T
  unsigned short* wCvTT[4]; unsigned short* wCvBT[4];
  for (int i = 0; i < 4; ++i) {
    wCvTT[i] = (unsigned short*)alloc(65536 * 2);         // 256x256^T
    wCvBT[i] = (unsigned short*)alloc(256 * 128 * 2);     // 128x256 -> [256][128]
  }

  auto castT = [&](const float* s, unsigned short* d, int K, int Nc) {
    long long n = (long long)K * Nc;
    cast_t_kernel<<<(unsigned)((n + 255) / 256), 256, 0, stream>>>(s, d, K, Nc);
  };
  castT(ef_W, wEfT, DEI, DE2);
  castT(hK, wHKT, D_, D_); castT(hQ, wHQT, D_, D_); castT(hV, wHVT, D_, D_);
  castT(gK, wGKT, D_, D_); castT(gQ, wGQT, D_, D_); castT(gV, wGVT, D_, D_);
  castT(last_W, wLaTT, D_, D_); castT(last_W + 256 * 256, wLaBT, D_, D_);
  for (int i = 0; i < 4; ++i) {
    castT(conv_W[i], wCvTT[i], D_, D_);
    castT(conv_W[i] + 256 * 256, wCvBT[i], DE2, D_);
  }
  cast_bf16_kernel<<<(unsigned)(E_ * DEI / 256), 256, 0, stream>>>(efeat, ebf, E_ * DEI);
  copy_cast_kernel<<<(unsigned)(N_ * D_ / 256), 256, 0, stream>>>(x_in, xcur, xbf, N_ * D_);

  // ef = relu(edge_features @ ef_W + ef_b)
  {
    dim3 g((unsigned)(E_ / 128), 1);
    gemm_bf16_kernel<GF_RELU><<<g, 256, 0, stream>>>(ebf, DEI, wEfT, DEI, efbuf, DE2,
                                                     nullptr, 0, ef_b, DEI);
  }
  hipMemsetAsync(efsum, 0, N_ * DE2 * 4, stream);
  hipMemsetAsync(cnt, 0, N_ * 4, stream);
  efscatter_kernel<<<(unsigned)E_, DE2, 0, stream>>>(efbuf, edst, efsum, cnt);
  cast_bf16_kernel<<<(unsigned)(N_ * DE2 / 256), 256, 0, stream>>>(efsum, efsumbf, N_ * DE2);

  dim3 gN((unsigned)(N_ / 128), D_ / 128);   // dense N-row, Nc=256
  dim3 gH(512, D_ / 128);                    // H/G-block GEMM, M=65536
  dim3 gS((unsigned)(N_ / 128), 4);          // scores, Nc=512

  for (int i = 0; i < 4; ++i) {
    gemm_bf16_kernel<0><<<gN, 256, 0, stream>>>(xbf, D_, wCvTT[i], D_, ybuf, D_,
                                                nullptr, 0, nullptr, D_);
    hipMemsetAsync(accb, 0, N_ * D_ * 4, stream);
    yscatter_kernel<<<(unsigned)E_, D_, 0, stream>>>(ybuf, esrc, edst, accb);
    gemm_bf16_kernel<GF_ACC><<<gN, 256, 0, stream>>>(efsumbf, DE2, wCvBT[i], DE2, accb, D_,
                                                     nullptr, 0, nullptr, DE2);
    conv_fin_kernel<<<(unsigned)N_, 256, 0, stream>>>(accb, cnt, conv_b[i], ybuf);
    gnorm_kernel<<<256, 256, 0, stream>>>(ybuf, gn_w, gn_b, gn_ms, xnew, xnewbf);

    if (i == 0) {
      copy_cast_kernel<<<(unsigned)(N_ * D_ / 256), 256, 0, stream>>>(xnew, xcur, xbf, N_ * D_);
    } else {
      gemm_bf16_kernel<GF_RELU | GF_IH><<<gH, 256, 0, stream>>>(xbf, D_, wHQT, D_, ctmp, D_, qbf, D_, nullptr, D_);
      gemm_bf16_kernel<GF_RELU | GF_IG><<<gH, 256, 0, stream>>>(xbf, D_, wGQT, D_, ctmp, D_, qbf, D_, nullptr, D_);
      gemm_bf16_kernel<GF_RELU | GF_IH><<<gH, 256, 0, stream>>>(xbf, D_, wHKT, D_, ctmp, D_, kbf, D_, nullptr, D_);
      gemm_bf16_kernel<GF_RELU | GF_IG><<<gH, 256, 0, stream>>>(xbf, D_, wGKT, D_, ctmp, D_, kbf, D_, nullptr, D_);
      gemm_bf16_kernel<GF_IH | GF_CBT><<<gH, 256, 0, stream>>>(xbf, D_, wHVT, D_, ctmp, D_, vT, (int)N_, nullptr, D_);
      gemm_bf16_kernel<GF_IG | GF_CBT><<<gH, 256, 0, stream>>>(xbf, D_, wGVT, D_, ctmp, D_, vT, (int)N_, nullptr, D_);
      // scores = q @ k_opp^T
      gemm_bf16_kernel<GF_OPPN><<<gS, 256, 0, stream>>>(qbf, D_, kbf, D_, scores, 512,
                                                        nullptr, 0, nullptr, D_);
      softmax_kernel<<<(unsigned)N_, 256, 0, stream>>>(scores, probs);
      // att = probs @ v_opp   (B = vT[d][node], opp offset on k)
      gemm_bf16_kernel<GF_OPPK><<<gN, 256, 0, stream>>>(probs, 512, vT, (int)N_, xcross, D_,
                                                        nullptr, 0, nullptr, 512);
      gnorm_kernel<<<256, 256, 0, stream>>>(xcross, gn_w, gn_b, gn_ms, xcrossn, xcnbf);
      gemm_bf16_kernel<0><<<gN, 256, 0, stream>>>(xnewbf, D_, wLaTT, D_, ctmp, D_,
                                                  nullptr, 0, nullptr, D_);
      gemm_bf16_kernel<GF_ACC | GF_RELU><<<gN, 256, 0, stream>>>(xcnbf, D_, wLaBT, D_, ctmp, D_,
                                                                 nullptr, 0, last_b, D_);
      mix_kernel<<<(unsigned)(N_ * D_ / 256), 256, 0, stream>>>(xcur, ctmp, xbf, N_ * D_);
    }
  }

  segmean_kernel<<<256, 256, 0, stream>>>(xcur, xout);
  float* out_scores = (float*)d_out;
  float* out_x  = out_scores + 256;
  float* out_x1 = out_x + N_ * D_;
  float* out_x2 = out_x1 + (long long)G_ * D_;
  final_kernel<<<G_, 256, 0, stream>>>(xout, fin1_W, fin1_b, fin2_W, fin2_b,
                                       out_scores, out_x1, out_x2);
  copy_kernel<<<(unsigned)(N_ * D_ / 256), 256, 0, stream>>>(xcur, out_x, N_ * D_);
}
